// VEncoderLayerPyG_68951404970536
// MI455X (gfx1250) — matrix operand order repaired and checked by reading, not compiled, not run on hardware
//
#include <hip/hip_runtime.h>
#include <hip/hip_bf16.h>

typedef __attribute__((ext_vector_type(16))) __bf16 v16bf;
typedef __attribute__((ext_vector_type(8)))  __bf16 v8bf;
typedef __attribute__((ext_vector_type(8)))  float  v8f;
typedef __attribute__((ext_vector_type(4)))  float  v4f;

union V16U { v16bf v; v8bf h[2]; };

constexpr int N_NODES = 50000;
constexpr int N_EDGES = 800000;
constexpr int D = 128;   // embed dim
constexpr int H = 8;     // heads
constexpr int F = 512;   // ff hidden
constexpr int EPLUS = N_EDGES + N_NODES;   // edges + self loops

// ---------------------------------------------------------------- WMMA utils

__device__ inline v8f wmma_bf16(v16bf a, v16bf b, v8f c) {
  // (neg_a, A, neg_b, B, c_mod, C, reuse_a, reuse_b)
  return __builtin_amdgcn_wmma_f32_16x16x32_bf16(false, a, false, b, (short)0, c,
                                                 false, false);
}

// A fragment: 16x32 bf16 subtile, K chunk at k0.
// lanes 0-15 row L: K=[k0..k0+7],[k0+16..k0+23]; lanes 16-31 row L-16: +8 each.
__device__ inline v16bf load_A_frag(const __bf16* base, int rowStride, int k0, int lane) {
  const __bf16* p = base + (lane & 15) * rowStride + k0 + ((lane & 16) ? 8 : 0);
  V16U u;
  u.h[0] = *(const v8bf*)(p);
  u.h[1] = *(const v8bf*)(p + 16);
  return u.v;
}

// B fragment: 32x16 where B[k][n] = W[n][k], W row-major [Nout, K].
// lanes 0-15: col n0+L, K=[k0..k0+7],[k0+16..k0+23]; lanes 16-31: col n0+L-16, +8 each.
__device__ inline v16bf load_B_frag(const __bf16* W, int ldk, int n0, int k0, int lane) {
  const __bf16* p = W + (long)(n0 + (lane & 15)) * ldk + k0 + ((lane & 16) ? 8 : 0);
  V16U u;
  u.h[0] = *(const v8bf*)(p);
  u.h[1] = *(const v8bf*)(p + 16);
  return u.v;
}

__device__ inline float wave_sum(float v) {
  for (int o = 16; o > 0; o >>= 1) v += __shfl_xor(v, o, 32);
  return v;
}

__device__ inline void atomicMaxF(float* addr, float v) {
  // sign-aware int trick; buffer initialized to -inf
  if (v >= 0.f) atomicMax((int*)addr, __float_as_int(v));
  else          atomicMin((unsigned int*)addr, __float_as_uint(v));
}

__device__ inline float leaky(float v) { return (v > 0.f) ? v : 0.2f * v; }

// ------------------------------------------------------------------ kernels

__global__ void __launch_bounds__(256) k_init(float* gat, float* m, float* s) {
  int i = blockIdx.x * 256 + threadIdx.x;
  if (i < N_NODES * D) gat[i] = 0.f;
  if (i < N_NODES * H) { m[i] = -__builtin_inff(); s[i] = 0.f; }
}

__global__ void __launch_bounds__(256) k_wconv(const float* __restrict__ Wg,
                                               const float* __restrict__ W1,
                                               const float* __restrict__ W2,
                                               __bf16* Wgb, __bf16* W1b, __bf16* W2b) {
  int i = blockIdx.x * 256 + threadIdx.x;
  if (i < D * D) Wgb[i] = (__bf16)Wg[i];
  if (i < F * D) { W1b[i] = (__bf16)W1[i]; W2b[i] = (__bf16)W2[i]; }
}

// xl = x @ W_gat^T   (block: 128 rows; 8 waves each own a 16-col strip)
__global__ void __launch_bounds__(256) k_gemm_xl(const float* __restrict__ x,
                                                 const __bf16* __restrict__ Wb,
                                                 float* __restrict__ xl) {
  __shared__ __attribute__((aligned(16))) __bf16 sX[128 * 136];
  const int tid = threadIdx.x;
  const int m0 = blockIdx.x * 128;
  for (int i = 0; i < 64; ++i) {
    int idx = tid + i * 256;
    int row = idx >> 7, col = idx & 127;
    int gr = m0 + row;
    float v = (gr < N_NODES) ? x[(long)gr * D + col] : 0.f;
    sX[row * 136 + col] = (__bf16)v;
  }
  __syncthreads();
  const int w = tid >> 5, lane = tid & 31;
  const int n0 = w * 16;
  const int col = n0 + (lane & 15);
  const int rbase = (lane & 16) ? 8 : 0;
  // B fragments identical for all 8 M-tiles: hoist into registers
  v16bf bf[4];
  for (int kk = 0; kk < 4; ++kk) bf[kk] = load_B_frag(Wb, 128, n0, kk * 32, lane);
  for (int mt = 0; mt < 8; ++mt) {
    int rowTile = m0 + mt * 16;           // N_NODES % 16 == 0 -> tile all-or-nothing
    if (rowTile >= N_NODES) break;        // wave-uniform
    v8f acc = {};
    const __bf16* Abase = sX + mt * 16 * 136;
    for (int kk = 0; kk < 4; ++kk)
      acc = wmma_bf16(load_A_frag(Abase, 136, kk * 32, lane), bf[kk], acc);
    float* p = xl + (long)(rowTile + rbase) * D + col;
    for (int r = 0; r < 8; ++r) p[(long)r * D] = acc[r];   // unguarded
  }
}

// per (node, head) attention coefficients
__global__ void __launch_bounds__(256) k_asad(const float* __restrict__ xl,
                                              const float* __restrict__ att_s,
                                              const float* __restrict__ att_d,
                                              float* __restrict__ a_s,
                                              float* __restrict__ a_d) {
  int i = blockIdx.x * 256 + threadIdx.x;
  if (i >= N_NODES * H) return;
  int n = i >> 3, h = i & 7;
  const float* row = xl + (long)n * D + h * 16;
  const float* as = att_s + h * 16;
  const float* ad = att_d + h * 16;
  float ss = 0.f, sd = 0.f;
  for (int c = 0; c < 16; ++c) { ss += row[c] * as[c]; sd += row[c] * ad[c]; }
  a_s[i] = ss; a_d[i] = sd;
}

__device__ inline void edge_sd(const long long* ei, int e, int& s, int& d) {
  if (e < N_EDGES) { s = (int)ei[e]; d = (int)ei[N_EDGES + e]; }
  else             { s = d = e - N_EDGES; }
}

// one thread per edge: vector-gather both 8-head coefficient rows, 8 atomics
__global__ void __launch_bounds__(256) k_edge_max(const long long* __restrict__ ei,
                                                  const float* __restrict__ a_s,
                                                  const float* __restrict__ a_d,
                                                  float* __restrict__ m) {
  int e = blockIdx.x * 256 + threadIdx.x;
  if (e >= EPLUS) return;
  int s, d;
  edge_sd(ei, e, s, d);
  v4f as0 = *(const v4f*)(a_s + (long)s * H);
  v4f as1 = *(const v4f*)(a_s + (long)s * H + 4);
  v4f ad0 = *(const v4f*)(a_d + (long)d * H);
  v4f ad1 = *(const v4f*)(a_d + (long)d * H + 4);
  float* mp = m + (long)d * H;
  for (int h = 0; h < 4; ++h) atomicMaxF(mp + h,     leaky(as0[h] + ad0[h]));
  for (int h = 0; h < 4; ++h) atomicMaxF(mp + 4 + h, leaky(as1[h] + ad1[h]));
}

__global__ void __launch_bounds__(256) k_edge_sum(const long long* __restrict__ ei,
                                                  const float* __restrict__ a_s,
                                                  const float* __restrict__ a_d,
                                                  const float* __restrict__ m,
                                                  float* __restrict__ sums) {
  int e = blockIdx.x * 256 + threadIdx.x;
  if (e >= EPLUS) return;
  int s, d;
  edge_sd(ei, e, s, d);
  v4f as0 = *(const v4f*)(a_s + (long)s * H);
  v4f as1 = *(const v4f*)(a_s + (long)s * H + 4);
  v4f ad0 = *(const v4f*)(a_d + (long)d * H);
  v4f ad1 = *(const v4f*)(a_d + (long)d * H + 4);
  v4f m0v = *(const v4f*)(m + (long)d * H);
  v4f m1v = *(const v4f*)(m + (long)d * H + 4);
  float* sp = sums + (long)d * H;
  for (int h = 0; h < 4; ++h) atomicAdd(sp + h,     expf(leaky(as0[h] + ad0[h]) - m0v[h]));
  for (int h = 0; h < 4; ++h) atomicAdd(sp + 4 + h, expf(leaky(as1[h] + ad1[h]) - m1v[h]));
}

// one wave32 per edge: alpha computed on lanes 0-7 (coalesced 32B gathers),
// broadcast via shfl; 4 channels/lane atomic scatter into L2-resident acc
__global__ void __launch_bounds__(256) k_edge_scatter(const long long* __restrict__ ei,
                                                      const float* __restrict__ a_s,
                                                      const float* __restrict__ a_d,
                                                      const float* __restrict__ m,
                                                      const float* __restrict__ sums,
                                                      const float* __restrict__ xl,
                                                      float* __restrict__ gat) {
  int wave = (blockIdx.x * 256 + threadIdx.x) >> 5;
  int lane = threadIdx.x & 31;
  if (wave >= EPLUS) return;
  int s, d;
  edge_sd(ei, wave, s, d);
  float alpha = 0.f;
  if (lane < H) {
    float v = leaky(a_s[(long)s * H + lane] + a_d[(long)d * H + lane]);
    alpha = expf(v - m[(long)d * H + lane]) / (sums[(long)d * H + lane] + 1e-16f);
  }
  alpha = __shfl(alpha, lane >> 2, 32);    // head = lane/4 (4 channels per lane)
  v4f xv = *(const v4f*)(xl + (long)s * D + lane * 4);
  float* gp = gat + (long)d * D + lane * 4;
  atomicAdd(gp + 0, xv[0] * alpha);
  atomicAdd(gp + 1, xv[1] * alpha);
  atomicAdd(gp + 2, xv[2] * alpha);
  atomicAdd(gp + 3, xv[3] * alpha);
}

// h = LayerNorm(x + gat + bias_gat) ; one wave per node
__global__ void __launch_bounds__(256) k_ln1(const float* __restrict__ x,
                                             const float* __restrict__ gat,
                                             const float* __restrict__ bias,
                                             const float* __restrict__ g1,
                                             const float* __restrict__ be1,
                                             float* __restrict__ h) {
  int gtid = blockIdx.x * 256 + threadIdx.x;
  int node = gtid >> 5, lane = gtid & 31;
  if (node >= N_NODES) return;
  long base = (long)node * D + lane * 4;
  v4f xv = *(const v4f*)(x + base);
  v4f gv = *(const v4f*)(gat + base);
  v4f bv = *(const v4f*)(bias + lane * 4);
  v4f p; float s1 = 0.f;
  for (int j = 0; j < 4; ++j) { p[j] = xv[j] + gv[j] + bv[j]; s1 += p[j]; }
  s1 = wave_sum(s1);
  float mu = s1 * (1.f / 128.f);
  float s2 = 0.f;
  for (int j = 0; j < 4; ++j) { float dd = p[j] - mu; s2 += dd * dd; }
  s2 = wave_sum(s2);
  float inv = rsqrtf(s2 * (1.f / 128.f) + 1e-5f);
  v4f gw = *(const v4f*)(g1 + lane * 4);
  v4f bw = *(const v4f*)(be1 + lane * 4);
  float* hp = h + base;
  for (int j = 0; j < 4; ++j) hp[j] = (p[j] - mu) * inv * gw[j] + bw[j];
}

// fused FFN: hidden = relu(h@W1^T+b1); out = LN2(h + hidden@W2^T + b2)
__global__ void __launch_bounds__(256) k_ffn(const float* __restrict__ h,
                                             const __bf16* __restrict__ W1b,
                                             const float* __restrict__ b1,
                                             const __bf16* __restrict__ W2b,
                                             const float* __restrict__ b2,
                                             const float* __restrict__ g2,
                                             const float* __restrict__ be2,
                                             float* __restrict__ out) {
  __shared__ __attribute__((aligned(16))) __bf16 sA[16 * 136];     // h tile bf16
  __shared__ __attribute__((aligned(16))) float  sH[16 * 132];     // h tile f32
  __shared__ __attribute__((aligned(16))) __bf16 sHid[16 * 520];   // relu hidden
  __shared__ __attribute__((aligned(16))) float  sP[16 * 132];     // pre-LN2
  const int tid = threadIdx.x;
  const int m0 = blockIdx.x * 16;           // 50000 % 16 == 0, no guards
  for (int i = 0; i < 8; ++i) {
    int idx = tid + i * 256;
    int row = idx >> 7, col = idx & 127;
    float v = h[(long)(m0 + row) * D + col];
    sA[row * 136 + col] = (__bf16)v;
    sH[row * 132 + col] = v;
  }
  __syncthreads();
  const int w = tid >> 5, lane = tid & 31;
  const int rbase = (lane & 16) ? 8 : 0;
  // GEMM1 (16x512) + bias + ReLU -> bf16 LDS.  A fragments shared by all 4 N-tiles.
  v16bf af[4];
  for (int kk = 0; kk < 4; ++kk) af[kk] = load_A_frag(sA, 136, kk * 32, lane);
  for (int t = 0; t < 4; ++t) {
    int n0 = (w * 4 + t) * 16;
    v8f acc = {};
    for (int kk = 0; kk < 4; ++kk)
      acc = wmma_bf16(af[kk], load_B_frag(W1b, 128, n0, kk * 32, lane), acc);
    int col = n0 + (lane & 15);
    float bb = b1[col];
    for (int r = 0; r < 8; ++r) {
      float v = acc[r] + bb;
      sHid[(rbase + r) * 520 + col] = (__bf16)(v > 0.f ? v : 0.f);
    }
  }
  __syncthreads();
  // GEMM2 (16x128) + bias + residual -> f32 LDS
  {
    int n0 = w * 16;
    v8f acc = {};
    for (int k0 = 0; k0 < 512; k0 += 32) {
      v16bf a = load_A_frag(sHid, 520, k0, lane);
      v16bf b = load_B_frag(W2b, 512, n0, k0, lane);
      acc = wmma_bf16(a, b, acc);
    }
    int col = n0 + (lane & 15);
    float bb = b2[col];
    for (int r = 0; r < 8; ++r) {
      int row = rbase + r;
      sP[row * 132 + col] = acc[r] + bb + sH[row * 132 + col];
    }
  }
  __syncthreads();
  // LN2: 8 waves x 2 rows
  for (int rr = w * 2; rr < w * 2 + 2; ++rr) {
    v4f v = *(const v4f*)(sP + rr * 132 + lane * 4);
    float s1 = wave_sum(v[0] + v[1] + v[2] + v[3]);
    float mu = s1 * (1.f / 128.f);
    float s2 = 0.f;
    for (int j = 0; j < 4; ++j) { float dd = v[j] - mu; s2 += dd * dd; }
    s2 = wave_sum(s2);
    float inv = rsqrtf(s2 * (1.f / 128.f) + 1e-5f);
    v4f gw = *(const v4f*)(g2 + lane * 4);
    v4f bw = *(const v4f*)(be2 + lane * 4);
    float* op = out + (long)(m0 + rr) * D + lane * 4;
    for (int j = 0; j < 4; ++j) op[j] = (v[j] - mu) * inv * gw[j] + bw[j];
  }
}

// ------------------------------------------------------------------ launch

extern "C" void kernel_launch(void* const* d_in, const int* in_sizes, int n_in,
                              void* d_out, int out_size, void* d_ws, size_t ws_size,
                              hipStream_t stream) {
  (void)in_sizes; (void)n_in; (void)out_size; (void)ws_size;
  const float*     x      = (const float*)d_in[0];
  const long long* ei     = (const long long*)d_in[1];   // int64 [2, E]
  const float*     Wg     = (const float*)d_in[2];
  const float*     att_s  = (const float*)d_in[3];
  const float*     att_d  = (const float*)d_in[4];
  const float*     bias_g = (const float*)d_in[5];
  const float*     W1     = (const float*)d_in[6];
  const float*     b1     = (const float*)d_in[7];
  const float*     W2     = (const float*)d_in[8];
  const float*     b2     = (const float*)d_in[9];
  const float*     g1     = (const float*)d_in[10];
  const float*     be1    = (const float*)d_in[11];
  const float*     g2     = (const float*)d_in[12];
  const float*     be2    = (const float*)d_in[13];
  float* out = (float*)d_out;

  // workspace carve (floats)
  float*  xl   = (float*)d_ws;                      // N*D
  float*  gat  = xl + (size_t)N_NODES * D;          // N*D
  float*  a_s  = gat + (size_t)N_NODES * D;         // N*H
  float*  a_d  = a_s + (size_t)N_NODES * H;
  float*  mbuf = a_d + (size_t)N_NODES * H;
  float*  sbuf = mbuf + (size_t)N_NODES * H;
  __bf16* Wgb  = (__bf16*)(sbuf + (size_t)N_NODES * H);
  __bf16* W1b  = Wgb + D * D;
  __bf16* W2b  = W1b + F * D;
  float*  hbuf = xl;  // reuse: xl dead after scatter

  k_init<<<(N_NODES * D) / 256, 256, 0, stream>>>(gat, mbuf, sbuf);
  k_wconv<<<(F * D) / 256, 256, 0, stream>>>(Wg, W1, W2, Wgb, W1b, W2b);
  k_gemm_xl<<<(N_NODES + 127) / 128, 256, 0, stream>>>(x, Wgb, xl);
  k_asad<<<(N_NODES * H + 255) / 256, 256, 0, stream>>>(xl, att_s, att_d, a_s, a_d);
  k_edge_max<<<(EPLUS + 255) / 256, 256, 0, stream>>>(ei, a_s, a_d, mbuf);
  k_edge_sum<<<(EPLUS + 255) / 256, 256, 0, stream>>>(ei, a_s, a_d, mbuf, sbuf);
  k_edge_scatter<<<EPLUS / 8, 256, 0, stream>>>(ei, a_s, a_d, mbuf, sbuf, xl, gat);
  k_ln1<<<(N_NODES + 7) / 8, 256, 0, stream>>>(x, gat, bias_g, g1, be1, hbuf);
  k_ffn<<<N_NODES / 16, 256, 0, stream>>>(hbuf, W1b, b1, W2b, b2, g2, be2, out);
}
